// DilationSpconv_7370163880515
// MI455X (gfx1250) — matrix operand order repaired and checked by reading, not compiled, not run on hardware
//
#include <hip/hip_runtime.h>
#include <hip/hip_bf16.h>

typedef __attribute__((ext_vector_type(16))) _Float16 v16h;
typedef __attribute__((ext_vector_type(8)))  _Float16 v8h;
typedef __attribute__((ext_vector_type(8)))  float    v8f;

#define XLIM 352
#define YLIM 96

// ---------------------------------------------------------------------------
// Prep 1: convert f32 features -> f16 buffer, and zero the pad row (row N) of
// both ping-pong buffers (index sentinel N selects the zero feature).
// ---------------------------------------------------------------------------
__global__ void cvt_feat_kernel(const float* __restrict__ feat,
                                _Float16* __restrict__ bufA,
                                _Float16* __restrict__ bufB, int total /* n*64 */) {
    int i = blockIdx.x * blockDim.x + threadIdx.x;
    if (i < total) {
        bufA[i] = (_Float16)feat[i];
    } else if (i < total + 64) {
        bufA[i] = (_Float16)0.f;   // pad row N of bufA
        bufB[i] = (_Float16)0.f;   // pad row N of bufB
    }
}

// ---------------------------------------------------------------------------
// Prep 2: pre-swizzle W[9,64,64] (f32, row-major [k,c,d]) for all 3 layers into
// f16 B-fragments in a contiguous-K B-matrix lane layout (same K order as loadA's A fragments):
//   fragment id f = ((layer*9 + k)*2 + kc)*4 + nt   (kc = K-chunk of 32, nt = N-tile)
//   lane l (0..31): N = nt*16 + (l&15); halves j=0..15 hold K = (l<16?0:16)+j
// Each fragment = 32 lanes * 16 halves = 512 halves; one contiguous 32B load/lane.
// ---------------------------------------------------------------------------
__global__ void swz_w_kernel(const float* __restrict__ W1,
                             const float* __restrict__ W2,
                             const float* __restrict__ W3,
                             _Float16* __restrict__ swz) {
    int t = blockIdx.x * blockDim.x + threadIdx.x;   // 3*9*2*4*32 = 6912 threads
    if (t >= 6912) return;
    int lane = t & 31;
    int f    = t >> 5;                // 0..215
    int nt    = f & 3;
    int kc    = (f >> 2) & 1;
    int k     = (f >> 3) % 9;
    int layer = f / 72;
    const float* W = (layer == 0) ? W1 : (layer == 1) ? W2 : W3;
    int d     = nt * 16 + (lane & 15);
    int kbase = (lane < 16) ? 0 : 16;
    _Float16* o = swz + (size_t)t * 16;
#pragma unroll
    for (int j = 0; j < 16; ++j) {
        int c = kc * 32 + kbase + j;
        o[j] = (_Float16)W[(size_t)k * 4096 + (size_t)c * 64 + d];
    }
}

// ---------------------------------------------------------------------------
// Prep 3: fold BatchNorm into per-channel scale/shift: s = g*rsqrt(v+eps),
// b' = b - m*s, for all 3 layers (layer-major, 64 each).
// ---------------------------------------------------------------------------
__global__ void bn_prep_kernel(const float* g1, const float* b1, const float* m1, const float* v1,
                               const float* g2, const float* b2, const float* m2, const float* v2,
                               const float* g3, const float* b3, const float* m3, const float* v3,
                               float* __restrict__ s, float* __restrict__ bs) {
    int i = blockIdx.x * blockDim.x + threadIdx.x;
    if (i >= 192) return;
    int layer = i >> 6, c = i & 63;
    const float* g = (layer == 0) ? g1 : (layer == 1) ? g2 : g3;
    const float* b = (layer == 0) ? b1 : (layer == 1) ? b2 : b3;
    const float* m = (layer == 0) ? m1 : (layer == 1) ? m2 : m3;
    const float* v = (layer == 0) ? v1 : (layer == 1) ? v2 : v3;
    float sc = g[c] * rsqrtf(v[c] + 1e-5f);
    s[i]  = sc;
    bs[i] = b[c] - m[c] * sc;
}

// ---------------------------------------------------------------------------
// Prep 4: coordinate mask per voxel row.
// ---------------------------------------------------------------------------
__global__ void mask_kernel(const int* __restrict__ coor, float* __restrict__ maskf, int n) {
    int i = blockIdx.x * blockDim.x + threadIdx.x;
    if (i >= n) return;
    int x = coor[i * 3 + 1];
    int y = coor[i * 3 + 2];
    maskf[i] = (x > -XLIM && x <= XLIM && y > -YLIM && y <= YLIM) ? 1.f : 0.f;
}

// ---------------------------------------------------------------------------
// Core: one wave32 per TWO 16-row bands (32 rows), full 64 output channels.
// B fragments for the whole layer staged in LDS (73.7 KB) by the block.
// Per (k, kc) step: 4 B fragments from LDS, 2 pipelined A fragments from
// global gather, 8 independent v_wmma_f32_16x16x32_f16 (distinct accums).
// A fragments for step s+1 are prefetched before the WMMAs of step s.
// ---------------------------------------------------------------------------
template <bool LAST>
__global__ __launch_bounds__(256) void sconv_kernel(
    const _Float16* __restrict__ in,      // (n+1) x 64 f16, row n == zeros
    const int* __restrict__ kin,          // [9, n]
    const _Float16* __restrict__ swz,     // this layer's 72 fragments (global)
    const float* __restrict__ s,          // [64]
    const float* __restrict__ bs,         // [64]
    void* __restrict__ outv,              // f16 (n x 64) or f32 (n x 64)
    const float* __restrict__ maskf,      // [n] (LAST only)
    int n) {
    __shared__ alignas(16) _Float16 smem[36864];   // 72 frags * 512 halves

    // Cooperative LDS fill: 73728 B = 4608 x 16B chunks.
    {
        const float4* src = (const float4*)swz;
        float4*       dst = (float4*)smem;
        for (int i = threadIdx.x; i < 4608; i += 256) dst[i] = src[i];
    }
    __syncthreads();

    int wave = (blockIdx.x * blockDim.x + threadIdx.x) >> 5;
    int lane = threadIdx.x & 31;
    int nb   = n >> 4;
    int band0 = wave * 2, band1 = band0 + 1;
    if (band0 >= nb) return;                 // whole-wave uniform exit
    bool has1 = band1 < nb;
    int  row0 = band0 << 4;
    int  row1 = (has1 ? band1 : band0) << 4; // duplicate work if odd band count
    int  lrow = lane & 15;
    int  koff = (lane >= 16) ? 8 : 0;

    // Hoist all gather indices out of the hot loop.
    int idx0[9], idx1[9];
#pragma unroll
    for (int k = 0; k < 9; ++k) {
        idx0[k] = kin[(size_t)k * n + row0 + lrow];
        idx1[k] = kin[(size_t)k * n + row1 + lrow];
    }

    v8f acc[8];                              // [band*4 + nt]
#pragma unroll
    for (int i = 0; i < 8; ++i) acc[i] = v8f{};

    auto loadA = [&](int idx, int kc) {
        const _Float16* p = in + (size_t)idx * 64 + kc * 32 + koff * 2;
        v8h lo = *(const v8h*)p;             // K0..K0+7
        v8h hi = *(const v8h*)(p + 8);       // K0+8..K0+15
        v16h a;
#pragma unroll
        for (int j = 0; j < 8; ++j) { a[j] = lo[j]; a[8 + j] = hi[j]; }
        return a;
    };

    // 2-stage software pipeline over the 18 (k, kc) steps.
    v16h a0 = loadA(idx0[0], 0);
    v16h a1 = loadA(idx1[0], 0);
#pragma unroll
    for (int step = 0; step < 18; ++step) {
        const int k = step >> 1, kc = step & 1;
        v16h a0n = a0, a1n = a1;
        if (step < 17) {
            const int kn = (step + 1) >> 1, kcn = (step + 1) & 1;
            a0n = loadA(idx0[kn], kcn);
            a1n = loadA(idx1[kn], kcn);
        }
        const _Float16* wb = smem + (size_t)((k * 2 + kc) * 4) * 512 + lane * 16;
#pragma unroll
        for (int nt = 0; nt < 4; ++nt) {
            v16h b = *(const v16h*)(wb + nt * 512);
            acc[nt] = __builtin_amdgcn_wmma_f32_16x16x32_f16(
                false, a0, false, b, (short)0, acc[nt], false, false);
            acc[4 + nt] = __builtin_amdgcn_wmma_f32_16x16x32_f16(
                false, a1, false, b, (short)0, acc[4 + nt], false, false);
        }
        a0 = a0n; a1 = a1n;
    }

    // Epilogue. D layout: lane -> N = nt*16 + (lane&15); VGPR r -> M = r (+8 if lane>=16).
    float mrow0[8], mrow1[8];
    if (LAST) {
#pragma unroll
        for (int r = 0; r < 8; ++r) {
            mrow0[r] = maskf[row0 + r + koff];
            mrow1[r] = maskf[row1 + r + koff];
        }
    }
#pragma unroll
    for (int nt = 0; nt < 4; ++nt) {
        int   d  = nt * 16 + lrow;
        float sv = s[d];
        float bv = bs[d];
#pragma unroll
        for (int r = 0; r < 8; ++r) {
            float v0 = fmaxf(acc[nt][r] * sv + bv, 0.f);
            int   r0 = row0 + r + koff;
            if (LAST) {
                ((float*)outv)[(size_t)r0 * 64 + d] = v0 * mrow0[r];
            } else {
                ((_Float16*)outv)[(size_t)r0 * 64 + d] = (_Float16)v0;
            }
        }
        if (has1) {
#pragma unroll
            for (int r = 0; r < 8; ++r) {
                float v1 = fmaxf(acc[4 + nt][r] * sv + bv, 0.f);
                int   r1 = row1 + r + koff;
                if (LAST) {
                    ((float*)outv)[(size_t)r1 * 64 + d] = v1 * mrow1[r];
                } else {
                    ((_Float16*)outv)[(size_t)r1 * 64 + d] = (_Float16)v1;
                }
            }
        }
    }
}

// ---------------------------------------------------------------------------
extern "C" void kernel_launch(void* const* d_in, const int* in_sizes, int n_in,
                              void* d_out, int out_size, void* d_ws, size_t ws_size,
                              hipStream_t stream) {
    const float* feat = (const float*)d_in[0];
    const int*   coor = (const int*)d_in[1];
    const int*   kin  = (const int*)d_in[2];
    const float* W1 = (const float*)d_in[3];
    const float* g1 = (const float*)d_in[4];
    const float* b1 = (const float*)d_in[5];
    const float* m1 = (const float*)d_in[6];
    const float* v1 = (const float*)d_in[7];
    const float* W2 = (const float*)d_in[8];
    const float* g2 = (const float*)d_in[9];
    const float* b2 = (const float*)d_in[10];
    const float* m2 = (const float*)d_in[11];
    const float* v2 = (const float*)d_in[12];
    const float* W3 = (const float*)d_in[13];
    const float* g3 = (const float*)d_in[14];
    const float* b3 = (const float*)d_in[15];
    const float* m3 = (const float*)d_in[16];
    const float* v3 = (const float*)d_in[17];

    const int n = in_sizes[0] / 64;          // 120000

    // Workspace carve-up (each region 256B-aligned).
    char*  ws       = (char*)d_ws;
    size_t bufBytes = (((size_t)(n + 1) * 64 * 2) + 255) & ~(size_t)255;
    _Float16* bufA  = (_Float16*)ws;
    _Float16* bufB  = (_Float16*)(ws + bufBytes);
    _Float16* swz   = (_Float16*)(ws + 2 * bufBytes);         // 3*72*512 halves
    size_t swzBytes = ((3u * 72u * 512u * 2u) + 255) & ~(size_t)255;
    float* sArr     = (float*)(ws + 2 * bufBytes + swzBytes); // 192 + 192 floats
    float* bsArr    = sArr + 192;
    float* maskf    = (float*)(ws + 2 * bufBytes + swzBytes + 2048);

    // Prep
    int tot = n * 64 + 64;
    cvt_feat_kernel<<<(tot + 255) / 256, 256, 0, stream>>>(feat, bufA, bufB, n * 64);
    swz_w_kernel<<<(6912 + 255) / 256, 256, 0, stream>>>(W1, W2, W3, swz);
    bn_prep_kernel<<<1, 192, 0, stream>>>(g1, b1, m1, v1, g2, b2, m2, v2, g3, b3, m3, v3,
                                          sArr, bsArr);
    mask_kernel<<<(n + 255) / 256, 256, 0, stream>>>(coor, maskf, n);

    // 3 conv layers: one wave per 32-row pair of bands, 8 waves per block.
    int nb     = n >> 4;                     // 16-row bands
    int nw     = (nb + 1) / 2;               // waves (2 bands each)
    int blocks = (nw + 7) / 8;
    sconv_kernel<false><<<blocks, 256, 0, stream>>>(bufA, kin, swz,          sArr,       bsArr,
                                                    (void*)bufB, nullptr, n);
    sconv_kernel<false><<<blocks, 256, 0, stream>>>(bufB, kin, swz + 36864,  sArr + 64,  bsArr + 64,
                                                    (void*)bufA, nullptr, n);
    sconv_kernel<true ><<<blocks, 256, 0, stream>>>(bufA, kin, swz + 73728,  sArr + 128, bsArr + 128,
                                                    d_out, maskf, n);
}
